// LSTMNetwork_9371618640375
// MI455X (gfx1250) — matrix operand order repaired and checked by reading, not compiled, not run on hardware
//
#include <hip/hip_runtime.h>
#include <hip/hip_bf16.h>
#include <stdint.h>

// ---------------------------------------------------------------------------
// Quantized LSTM for MI455X (gfx1250, wave32, WMMA).
//
// Sequential recurrence dominates (512 dependent steps); per-step work per
// wave is 12 k-iters x 4 WMMA(16x16x32 f16), inline-0 C for the first k-iter.
// Round 4: sched_group_barrier directives (DS-read x8 then WMMA x4 per k-iter)
// so each WMMA group consumes B fragments whose ds_loads were issued a full
// group earlier, instead of ds_load x2 -> s_wait_dscnt 0 -> wmma.
// ---------------------------------------------------------------------------

typedef __attribute__((ext_vector_type(16))) _Float16 v16h;
typedef __attribute__((ext_vector_type(8)))  _Float16 v8h;
typedef __attribute__((ext_vector_type(8)))  float    v8f;

#define B_   256
#define T_   512
#define I_   128
#define H_   256
#define G_   1024        // 4*H
#define NWG  16          // persistent workgroups (j-partition of H)
#define NTHR 512         // 16 wave32 per WG
#define JPW  16          // hidden units per WG
#define IP   (I_ + 8)    // padded LDS K-stride (68 dwords -> conflict free b128)
#define HP   (H_ + 8)    // 132 dwords, mod 64 == 4

#if defined(__has_builtin)
#if __has_builtin(__builtin_amdgcn_sched_group_barrier)
#define SCHED_DS_THEN_WMMA()                                   \
  do {                                                         \
    __builtin_amdgcn_sched_group_barrier(0x100, 8, 0); /*DS rd*/ \
    __builtin_amdgcn_sched_group_barrier(0x008, 4, 0); /*WMMA*/  \
  } while (0)
#define SCHED_WMMA_TAIL() __builtin_amdgcn_sched_group_barrier(0x008, 4, 0)
#endif
#endif
#ifndef SCHED_DS_THEN_WMMA
#define SCHED_DS_THEN_WMMA() do {} while (0)
#define SCHED_WMMA_TAIL()    do {} while (0)
#endif

__device__ __forceinline__ float quantf(float x) {
  // round to q8.8, clamp to int16 range / 256
  float q = __builtin_rintf(x * 256.0f) * 0.00390625f;
  return fminf(fmaxf(q, -128.0f), 127.99609375f);
}
__device__ __forceinline__ float hsig(float x) {
  return fminf(fmaxf(x * (1.0f / 6.0f) + 0.5f, 0.0f), 1.0f);
}
__device__ __forceinline__ float htanh(float x) {
  return fminf(fmaxf(x, -1.0f), 1.0f);
}

// ---- elementwise prep kernels ---------------------------------------------

__global__ void quant_x_kernel(const float* __restrict__ x,
                               _Float16* __restrict__ xq, int n) {
  int i = blockIdx.x * blockDim.x + threadIdx.x;
  if (i < n) xq[i] = (_Float16)quantf(x[i]);
}

__global__ void prep_kernel(const float* __restrict__ w_ih,
                            const float* __restrict__ w_hh,
                            const float* __restrict__ b_ih,
                            const float* __restrict__ b_hh,
                            const float* __restrict__ w_lin,
                            const float* __restrict__ b_lin,
                            _Float16* __restrict__ wih_h,
                            _Float16* __restrict__ whh_h,
                            float* __restrict__ bih_q,
                            float* __restrict__ bhh_q,
                            float* __restrict__ wl_q,
                            float* __restrict__ bl_q,
                            _Float16* __restrict__ hbuf,  // zero read-buffer 0
                            int* __restrict__ bar) {
  int i = blockIdx.x * blockDim.x + threadIdx.x;
  if (i < G_ * I_) wih_h[i] = (_Float16)quantf(w_ih[i]);
  if (i < G_ * H_) whh_h[i] = (_Float16)quantf(w_hh[i]);
  if (i < G_) { bih_q[i] = quantf(b_ih[i]); bhh_q[i] = quantf(b_hh[i]); }
  if (i < H_) wl_q[i] = quantf(w_lin[i]);
  if (i == 0) bl_q[0] = quantf(b_lin[0]);
  if (i < B_ * H_) hbuf[i] = (_Float16)0.0f;   // h0 = 0 (buffer 0)
  if (i < 2) bar[i] = 0;                       // grid barrier {cnt, gen}
}

// ---- WMMA fragment loaders (ISA 7.12.2 layouts, wave32) -------------------

// 16-bit A 16x32: lanes 0-15 row M=lane hold K {kb..kb+7, kb+16..kb+23},
// lanes 16-31 same rows hold K {kb+8..kb+15, kb+24..kb+31}.
__device__ __forceinline__ v16h load_a16(const _Float16* __restrict__ base,
                                         int ld, int row0, int kb, int lane) {
  int r  = row0 + (lane & 15);
  int kh = (lane >> 4) * 8;
  const _Float16* p = base + r * ld + kb + kh;
  v8h a0 = *(const v8h*)p;          // global_load_b128
  v8h a1 = *(const v8h*)(p + 16);   // global_load_b128
  v16h o;
#pragma unroll
  for (int i = 0; i < 8; ++i) { o[i] = a0[i]; o[i + 8] = a1[i]; }
  return o;
}

// 16-bit B 32x16: column N = lane&15; lanes 0-15 hold K {kb..kb+7, kb+16..kb+23},
// lanes 16-31 hold K {kb+8..kb+15, kb+24..kb+31} (matches A).
__device__ __forceinline__ v16h load_b16(const _Float16* __restrict__ colbase,
                                         int kb, int lane) {
  const _Float16* p = colbase + kb + (lane >> 4) * 8;
  v8h b0 = *(const v8h*)p;          // ds_load_b128
  v8h b1 = *(const v8h*)(p + 16);   // ds_load_b128
  v16h o;
#pragma unroll
  for (int i = 0; i < 8; ++i) { o[i] = b0[i]; o[i + 8] = b1[i]; }
  return o;
}

#define WMMA_F16(A, Bv, C) \
  __builtin_amdgcn_wmma_f32_16x16x32_f16(false, (A), false, (Bv), (short)0, (C), false, false)

// One GEMM phase: 16x64 tile, K-unrolled, fragment loads pipelined one
// k-iteration ahead of the WMMAs consuming them; sched_group_barriers pin the
// DS-read-x8 / WMMA-x4 interleave.
template <int K, int LDP>
__device__ __forceinline__ void gemm4(const _Float16* __restrict__ abase,
                                      int ald, int row0,
                                      const _Float16* __restrict__ lw,
                                      int n, int lane,
                                      v8f& a0, v8f& a1, v8f& a2, v8f& a3) {
  const _Float16* c0 = lw + (size_t)(n)      * LDP;
  const _Float16* c1 = lw + (size_t)(16 + n) * LDP;
  const _Float16* c2 = lw + (size_t)(32 + n) * LDP;
  const _Float16* c3 = lw + (size_t)(48 + n) * LDP;
  v16h af = load_a16(abase, ald, row0, 0, lane);
  v16h b0 = load_b16(c0, 0, lane);
  v16h b1 = load_b16(c1, 0, lane);
  v16h b2 = load_b16(c2, 0, lane);
  v16h b3 = load_b16(c3, 0, lane);
#pragma unroll
  for (int kb = 0; kb < K - 32; kb += 32) {
    // issue next iteration's loads before consuming current fragments
    v16h afn = load_a16(abase, ald, row0, kb + 32, lane);
    v16h b0n = load_b16(c0, kb + 32, lane);
    v16h b1n = load_b16(c1, kb + 32, lane);
    v16h b2n = load_b16(c2, kb + 32, lane);
    v16h b3n = load_b16(c3, kb + 32, lane);
    if (kb == 0) {
      const v8f z = {};
      a0 = WMMA_F16(af, b0, z);  a1 = WMMA_F16(af, b1, z);
      a2 = WMMA_F16(af, b2, z);  a3 = WMMA_F16(af, b3, z);
    } else {
      a0 = WMMA_F16(af, b0, a0); a1 = WMMA_F16(af, b1, a1);
      a2 = WMMA_F16(af, b2, a2); a3 = WMMA_F16(af, b3, a3);
    }
    SCHED_DS_THEN_WMMA();
    af = afn; b0 = b0n; b1 = b1n; b2 = b2n; b3 = b3n;
  }
  a0 = WMMA_F16(af, b0, a0); a1 = WMMA_F16(af, b1, a1);
  a2 = WMMA_F16(af, b2, a2); a3 = WMMA_F16(af, b3, a3);
  SCHED_WMMA_TAIL();
}

// ---- cooperative grid barrier (16 resident WGs) ---------------------------

__device__ __forceinline__ void grid_barrier(int* cnt, int* gen) {
  __builtin_amdgcn_fence(__ATOMIC_RELEASE, "agent");  // flush h stores to L2
  __syncthreads();
  if (threadIdx.x == 0) {
    int g = __hip_atomic_load(gen, __ATOMIC_RELAXED, __HIP_MEMORY_SCOPE_AGENT);
    int a = __hip_atomic_fetch_add(cnt, 1, __ATOMIC_ACQ_REL,
                                   __HIP_MEMORY_SCOPE_AGENT);
    if (a == NWG - 1) {
      __hip_atomic_store(cnt, 0, __ATOMIC_RELAXED, __HIP_MEMORY_SCOPE_AGENT);
      __hip_atomic_store(gen, g + 1, __ATOMIC_RELEASE,
                         __HIP_MEMORY_SCOPE_AGENT);
    } else {
      while (__hip_atomic_load(gen, __ATOMIC_ACQUIRE,
                               __HIP_MEMORY_SCOPE_AGENT) == g)
        __builtin_amdgcn_s_sleep(2);
    }
  }
  __syncthreads();
  __builtin_amdgcn_fence(__ATOMIC_ACQUIRE, "agent"); // invalidate for fresh h
}

// ---- persistent LSTM kernel ----------------------------------------------

__global__ __launch_bounds__(NTHR) void lstm_kernel(
    const _Float16* __restrict__ xq,       // [B][T][I] quantized f16
    const _Float16* __restrict__ wih_h,    // [G][I]
    const _Float16* __restrict__ whh_h,    // [G][H]
    const float* __restrict__ bih_q, const float* __restrict__ bhh_q,
    const float* __restrict__ wl_q, const float* __restrict__ bl_q,
    _Float16* __restrict__ hbuf,           // 2 x [B][H] double buffer
    int* __restrict__ bar,
    float* __restrict__ out) {
  __shared__ _Float16 lwih[64 * IP];       // 64 gate-cols x 128 K (padded)
  __shared__ _Float16 lwhh[64 * HP];       // 64 gate-cols x 256 K (padded)
  __shared__ float lbih[64], lbhh[64];

  const int tid = threadIdx.x;
  const int j0  = blockIdx.x * JPW;        // first hidden unit owned

  // Stage this WG's weight slice into LDS (columns K-contiguous).
  for (int idx = tid; idx < 64 * I_; idx += NTHR) {
    int col = idx >> 7, k = idx & (I_ - 1);
    int gc = (col >> 4) * H_ + j0 + (col & 15);
    lwih[col * IP + k] = wih_h[gc * I_ + k];
  }
  for (int idx = tid; idx < 64 * H_; idx += NTHR) {
    int col = idx >> 8, k = idx & (H_ - 1);
    int gc = (col >> 4) * H_ + j0 + (col & 15);
    lwhh[col * HP + k] = whh_h[gc * H_ + k];
  }
  if (tid < 64) {
    int gc = (tid >> 4) * H_ + j0 + (tid & 15);
    lbih[tid] = bih_q[gc];
    lbhh[tid] = bhh_q[gc];
  }
  __syncthreads();

  const int lane = tid & 31;
  const int wv   = tid >> 5;     // wave id 0..15 -> M tile (rows of B)
  const int row0 = wv * 16;
  const int n    = lane & 15;    // column within 16-wide N tile
  const int rhi  = 8 * (lane >> 4);
  const int arow = row0 + n;     // this lane's A-matrix row

  // Biases hoisted out of the t-loop (constant across steps).
  const float bi0 = lbih[n],      bi1 = lbih[16 + n],
              bi2 = lbih[32 + n], bi3 = lbih[48 + n];
  const float bh0 = lbhh[n],      bh1 = lbhh[16 + n],
              bh2 = lbhh[32 + n], bh3 = lbhh[48 + n];

  v8f cfrag = {};                // cell state 16x16 block, lives in VGPRs

  for (int t = 0; t < T_; ++t) {
    const _Float16* hread  = hbuf + (size_t)(t & 1) * (B_ * H_);
    _Float16*       hwrite = hbuf + (size_t)((t + 1) & 1) * (B_ * H_);

    // ---- phase 1: dot_x = xq_t @ wih^T  (bias folded into quant below) ----
    const _Float16* xbase = xq + (size_t)t * I_;  // + r*(T*I) -> x[r][t][.]
    v8f a0, a1, a2, a3;
    gemm4<I_, IP>(xbase, T_ * I_, row0, lwih, n, lane, a0, a1, a2, a3);

    v8f g0, g1, g2, g3;
#pragma unroll
    for (int e = 0; e < 8; ++e) {
      g0[e] = quantf(a0[e] + bi0); g1[e] = quantf(a1[e] + bi1);
      g2[e] = quantf(a2[e] + bi2); g3[e] = quantf(a3[e] + bi3);
    }

    // Prefetch next step's x rows while the h-phase runs (hides HBM latency).
    if (t + 1 < T_) {
      const _Float16* xn = xq + (size_t)(t + 1) * I_ + (size_t)arow * (T_ * I_);
      __builtin_prefetch(xn, 0, 3);
      __builtin_prefetch(xn + 64, 0, 3);
    }

    // ---- phase 2: dot_h = h @ whh^T ---------------------------------------
    gemm4<H_, HP>(hread, H_, row0, lwhh, n, lane, a0, a1, a2, a3);

    // ---- gates, cell/hidden update (C layout: e -> M=row0+e+rhi, N=n) -----
#pragma unroll
    for (int e = 0; e < 8; ++e) {
      float ig = quantf(g0[e] + quantf(a0[e] + bh0));
      float fg = quantf(g1[e] + quantf(a1[e] + bh1));
      float cg = quantf(g2[e] + quantf(a2[e] + bh2));
      float og = quantf(g3[e] + quantf(a3[e] + bh3));
      ig = quantf(hsig(ig));
      fg = quantf(hsig(fg));
      cg = quantf(htanh(cg));
      og = quantf(hsig(og));
      float cn = quantf(quantf(fg * cfrag[e]) + quantf(ig * cg));
      cfrag[e] = cn;
      float hn = quantf(og * quantf(htanh(cn)));
      hwrite[(row0 + e + rhi) * H_ + j0 + n] = (_Float16)hn;
    }

    grid_barrier(bar, bar + 1);
  }

  // ---- head: out[b] = quant(h_last[b] . wl_q + bl_q); h_last in buffer 0 --
  if (blockIdx.x == 0 && tid < B_) {
    const _Float16* hlast = hbuf;  // (T_ & 1) == 0 -> final write was buf 0
    float s = bl_q[0];
    for (int j = 0; j < H_; ++j) s += (float)hlast[tid * H_ + j] * wl_q[j];
    out[tid] = quantf(s);
  }
}

// ---------------------------------------------------------------------------

extern "C" void kernel_launch(void* const* d_in, const int* in_sizes, int n_in,
                              void* d_out, int out_size, void* d_ws,
                              size_t ws_size, hipStream_t stream) {
  (void)in_sizes; (void)n_in; (void)out_size; (void)ws_size;
  const float* x     = (const float*)d_in[0];
  const float* w_ih  = (const float*)d_in[1];
  const float* w_hh  = (const float*)d_in[2];
  const float* b_ih  = (const float*)d_in[3];
  const float* b_hh  = (const float*)d_in[4];
  const float* w_lin = (const float*)d_in[5];
  const float* b_lin = (const float*)d_in[6];

  char* ws = (char*)d_ws;
  size_t off = 0;
  auto alloc = [&](size_t bytes) -> char* {
    char* p = ws + off;
    off = (off + bytes + 255) & ~(size_t)255;
    return p;
  };
  _Float16* xq    = (_Float16*)alloc((size_t)B_ * T_ * I_ * 2);  // 32 MB
  _Float16* hbuf  = (_Float16*)alloc((size_t)2 * B_ * H_ * 2);
  _Float16* wih_h = (_Float16*)alloc((size_t)G_ * I_ * 2);
  _Float16* whh_h = (_Float16*)alloc((size_t)G_ * H_ * 2);
  float*    bih_q = (float*)alloc(G_ * 4);
  float*    bhh_q = (float*)alloc(G_ * 4);
  float*    wl_q  = (float*)alloc(H_ * 4);
  float*    bl_q  = (float*)alloc(4);
  int*      bar   = (int*)alloc(8);

  const int nx = B_ * T_ * I_;
  quant_x_kernel<<<(nx + 255) / 256, 256, 0, stream>>>(x, xq, nx);
  prep_kernel<<<(G_ * H_ + 255) / 256, 256, 0, stream>>>(
      w_ih, w_hh, b_ih, b_hh, w_lin, b_lin,
      wih_h, whh_h, bih_q, bhh_q, wl_q, bl_q, hbuf, bar);
  lstm_kernel<<<NWG, NTHR, 0, stream>>>(xq, wih_h, whh_h, bih_q, bhh_q,
                                        wl_q, bl_q, hbuf, bar, (float*)d_out);
}
